// GCNModel_83090437308937
// MI455X (gfx1250) — matrix operand order repaired and checked by reading, not compiled, not run on hardware
//
#include <hip/hip_runtime.h>
#include <hip/hip_bf16.h>

typedef __attribute__((ext_vector_type(16))) _Float16 v16h;
typedef __attribute__((ext_vector_type(8)))  _Float16 v8h;
typedef __attribute__((ext_vector_type(8)))  float    v8f;

// ---------------------------------------------------------------------------
// Utility kernels
// ---------------------------------------------------------------------------
__global__ void k_fill(float* __restrict__ p, float v, int n) {
    int t = blockIdx.x * blockDim.x + threadIdx.x;
    if (t < n) p[t] = v;
}

__global__ void k_deg_edges(const int* __restrict__ dst, float* __restrict__ deg, int E) {
    int t = blockIdx.x * blockDim.x + threadIdx.x;
    if (t < E) atomicAdd(&deg[dst[t]], 1.0f);
}

__global__ void k_dinv(float* __restrict__ deg, int n) {
    int t = blockIdx.x * blockDim.x + threadIdx.x;
    if (t < n) {
        float d = deg[t];
        deg[t] = (d > 0.0f) ? rsqrtf(d) : 0.0f;   // in-place deg -> dinv
    }
}

__global__ void k_cvt_f16(const float* __restrict__ src, _Float16* __restrict__ dstp, int n) {
    int t = blockIdx.x * blockDim.x + threadIdx.x;
    if (t < n) dstp[t] = (_Float16)src[t];
}

// W [K,N] f32 row-major  ->  Wt [N,K] f16 row-major (B transposed for WMMA)
__global__ void k_cvt_wT(const float* __restrict__ W, _Float16* __restrict__ Wt, int K, int N) {
    int t = blockIdx.x * blockDim.x + threadIdx.x;
    if (t < K * N) {
        int k = t / N, n = t % N;
        Wt[(size_t)n * K + k] = (_Float16)W[t];
    }
}

// ---------------------------------------------------------------------------
// WMMA GEMM: C[M,N] = A[M,K] * B[K,N], A f16 row-major, Bt = B^T f16 [N,K],
// C f32 row-major. One wave computes one 16x16 tile, K stepped by 32.
// ---------------------------------------------------------------------------
__global__ __launch_bounds__(256) void k_wmma_gemm(
    const _Float16* __restrict__ A,
    const _Float16* __restrict__ Bt,
    float* __restrict__ C,
    int M, int N, int K)
{
    const int wave   = (blockIdx.x * blockDim.x + threadIdx.x) >> 5;
    const int lane   = threadIdx.x & 31;
    const int ntiles = N >> 4;
    const int tiles  = (M >> 4) * ntiles;
    if (wave >= tiles) return;
    const int mt = wave / ntiles;
    const int nt = wave % ntiles;

    const int h16 = lane >> 4;   // half-wave: 0 = lanes 0-15, 1 = lanes 16-31
    const int l16 = lane & 15;

    // A fragment: row = mt*16 + l16; lanes 0-15 hold K 0..15,
    // lanes 16-31 hold K 16..31 within each 32-wide K step.
    const _Float16* arow = A  + (size_t)(mt * 16 + l16) * K + h16 * 16;
    // B fragment: col = nt*16 + l16; lanes 0-15 hold K 0..15, lanes 16-31
    // hold K 16..31 -> 16 contiguous halfs of the transposed weight row.
    const _Float16* brow = Bt + (size_t)(nt * 16 + l16) * K + h16 * 16;

    v8f acc = {};
    #pragma unroll
    for (int k0 = 0; k0 < 256; k0 += 32) {
        v8h alo = *(const v8h*)(arow + k0);        // K k0+off+0..7
        v8h ahi = *(const v8h*)(arow + k0 + 8);    // K k0+off+8..15
        v16h a  = __builtin_shufflevector(alo, ahi,
                     0,1,2,3,4,5,6,7,8,9,10,11,12,13,14,15);
        v16h b  = *(const v16h*)(brow + k0);       // 16 contiguous K values
        acc = __builtin_amdgcn_wmma_f32_16x16x32_f16(
                  false, a, false, b, (short)0, acc, false, false);
    }

    // C layout: lanes 0-15 -> rows 0..7, lanes 16-31 -> rows 8..15; col = l16.
    float* crow = C + (size_t)(mt * 16 + h16 * 8) * N + nt * 16 + l16;
    #pragma unroll
    for (int r = 0; r < 8; ++r) crow[(size_t)r * N] = acc[r];
}

// ---------------------------------------------------------------------------
// Edge scatter: agg[dst] += H[src] * dinv[src]*dinv[dst]  (float4 chunks)
// ---------------------------------------------------------------------------
__global__ void k_scatter(const float* __restrict__ H,
                          const float* __restrict__ dinv,
                          const int* __restrict__ src,
                          const int* __restrict__ dst,
                          float* __restrict__ agg,
                          int E, int D)
{
    const int chunks = D >> 2;
    long long t = (long long)blockIdx.x * blockDim.x + threadIdx.x;
    if (t >= (long long)E * chunks) return;
    int e = (int)(t / chunks);
    int c = (int)(t % chunks) << 2;
    int s = src[e], d = dst[e];
    float w = dinv[s] * dinv[d];
    const float4 h = *(const float4*)(H + (size_t)s * D + c);
    float* out = agg + (size_t)d * D + c;
    atomicAdd(out + 0, h.x * w);
    atomicAdd(out + 1, h.y * w);
    atomicAdd(out + 2, h.z * w);
    atomicAdd(out + 3, h.w * w);
}

// Layer-1 finalize: x = relu(agg + H*dinv^2 + b), emit f16 for next WMMA.
__global__ void k_finalize_relu_f16(const float* __restrict__ agg,
                                    const float* __restrict__ H,
                                    const float* __restrict__ dinv,
                                    const float* __restrict__ b,
                                    _Float16* __restrict__ Xh2,
                                    int Nn, int D)
{
    int t = blockIdx.x * blockDim.x + threadIdx.x;
    if (t >= Nn * D) return;
    int i = t / D, f = t % D;
    float di = dinv[i];
    float v = agg[t] + H[t] * di * di + b[f];
    Xh2[t] = (_Float16)fmaxf(v, 0.0f);
}

// Layer-2 finalize (in place on d_out): out = agg + H*dinv^2 + b.
__global__ void k_finalize_out(float* __restrict__ agg,
                               const float* __restrict__ H,
                               const float* __restrict__ dinv,
                               const float* __restrict__ b,
                               int Nn, int D)
{
    int t = blockIdx.x * blockDim.x + threadIdx.x;
    if (t >= Nn * D) return;
    int i = t / D, f = t % D;
    float di = dinv[i];
    agg[t] = agg[t] + H[t] * di * di + b[f];
}

// ---------------------------------------------------------------------------
extern "C" void kernel_launch(void* const* d_in, const int* in_sizes, int n_in,
                              void* d_out, int out_size, void* d_ws, size_t ws_size,
                              hipStream_t stream)
{
    const float* X  = (const float*)d_in[0];   // [N, 256]
    const int*   ei = (const int*)  d_in[1];   // [2, E]
    const float* W1 = (const float*)d_in[2];   // [256, 256]
    const float* b1 = (const float*)d_in[3];   // [256]
    const float* W2 = (const float*)d_in[4];   // [256, 128]
    const float* b2 = (const float*)d_in[5];   // [128]

    const int DIN = 256, DH = 256, DA = 128;
    const int Nn = in_sizes[0] / DIN;          // 20000
    const int E  = in_sizes[1] / 2;            // 320000
    const int* src = ei;
    const int* dst = ei + E;

    // --- carve workspace (256B aligned) ---
    char* w = (char*)d_ws;
    size_t off = 0;
    auto carve = [&](size_t bytes) -> void* {
        off = (off + 255) & ~(size_t)255;
        void* p = w + off;
        off += bytes;
        return p;
    };
    float*    dinv = (float*)   carve((size_t)Nn * sizeof(float));
    _Float16* Xh   = (_Float16*)carve((size_t)Nn * DIN * 2);
    _Float16* W1t  = (_Float16*)carve((size_t)DH * DIN * 2);   // [N=256][K=256]
    _Float16* W2t  = (_Float16*)carve((size_t)DA * DH  * 2);   // [N=128][K=256]
    float*    H1   = (float*)   carve((size_t)Nn * DH * 4);
    float*    AGG1 = (float*)   carve((size_t)Nn * DH * 4);
    _Float16* Xh2  = (_Float16*)carve((size_t)Nn * DH * 2);
    float*    H2   = (float*)   carve((size_t)Nn * DA * 4);
    (void)ws_size; (void)n_in;

    auto nb = [](long long n, int b) { return (unsigned)((n + b - 1) / b); };
    const int B = 256;

    // --- degrees -> dinv ---
    k_fill<<<nb(Nn, B), B, 0, stream>>>(dinv, 1.0f, Nn);          // self loop
    k_deg_edges<<<nb(E, B), B, 0, stream>>>(dst, dinv, E);
    k_dinv<<<nb(Nn, B), B, 0, stream>>>(dinv, Nn);

    // --- f16 conversions ---
    k_cvt_f16<<<nb((long long)Nn * DIN, B), B, 0, stream>>>(X, Xh, Nn * DIN);
    k_cvt_wT <<<nb((long long)DIN * DH, B), B, 0, stream>>>(W1, W1t, DIN, DH);
    k_cvt_wT <<<nb((long long)DH * DA, B), B, 0, stream>>>(W2, W2t, DH, DA);

    // --- layer 1: H1 = X @ W1 (WMMA) ---
    long long waves1 = (long long)(Nn / 16) * (DH / 16);
    k_wmma_gemm<<<nb(waves1 * 32, B), B, 0, stream>>>(Xh, W1t, H1, Nn, DH, DIN);

    // --- layer 1 aggregation ---
    k_fill<<<nb((long long)Nn * DH, B), B, 0, stream>>>(AGG1, 0.0f, Nn * DH);
    k_scatter<<<nb((long long)E * (DH / 4), B), B, 0, stream>>>(H1, dinv, src, dst, AGG1, E, DH);
    k_finalize_relu_f16<<<nb((long long)Nn * DH, B), B, 0, stream>>>(AGG1, H1, dinv, b1, Xh2, Nn, DH);

    // --- layer 2: H2 = X2 @ W2 (WMMA) ---
    long long waves2 = (long long)(Nn / 16) * (DA / 16);
    k_wmma_gemm<<<nb(waves2 * 32, B), B, 0, stream>>>(Xh2, W2t, H2, Nn, DA, DH);

    // --- layer 2 aggregation directly into d_out ---
    float* out = (float*)d_out;
    k_fill<<<nb((long long)Nn * DA, B), B, 0, stream>>>(out, 0.0f, Nn * DA);
    k_scatter<<<nb((long long)E * (DA / 4), B), B, 0, stream>>>(H2, dinv, src, dst, out, E, DA);
    k_finalize_out<<<nb((long long)Nn * DA, B), B, 0, stream>>>(out, H2, dinv, b2, Nn, DA);
}
